// SpeakerConditionedDecoder_51745765982604
// MI455X (gfx1250) — matrix operand order repaired and checked by reading, not compiled, not run on hardware
//
#include <hip/hip_runtime.h>
#include <hip/hip_bf16.h>

// ---------------- problem sizes ----------------
#define Bb     32
#define Ss     256
#define Tt     400
#define ENC    512
#define SPK    256
#define DEC    1024
#define ATT    128
#define MEL    80
#define PRE    256
#define PREOUT 128
#define Gg     4096           // 4*DEC
#define K1     1664           // (PREOUT+ENC) + DEC
#define K2     2048           // DEC + DEC
#define NKT1   52             // K1/32
#define NKT2   64             // K2/32
#define NSTRIP 256            // Gg/16
#define NWG    32
#define NT     256
#define GSZ    (NWG*NT)       // 8192 threads, 256 waves

typedef __attribute__((ext_vector_type(16))) __bf16 v16bf;
typedef __attribute__((ext_vector_type(8)))  float  v8f;

// ---------------- workspace layout (bytes) ----------------
#define ALG(x) (((x) + (size_t)255) & ~(size_t)255)
static constexpr size_t OFF_W1P  = 0;                                       // bf16 packed [NSTRIP][NKT1][32][16]
static constexpr size_t OFF_W2P  = ALG(OFF_W1P + (size_t)K1*Gg*2);
static constexpr size_t OFF_PK   = ALG(OFF_W2P + (size_t)K2*Gg*2);          // f32 [B][S][ATT]
static constexpr size_t OFF_SATT = ALG(OFF_PK  + (size_t)Bb*Ss*ATT*4);      // f32 [B][ATT]
static constexpr size_t OFF_SDEC = ALG(OFF_SATT+ (size_t)Bb*ATT*4);         // f32 [B][DEC]
static constexpr size_t OFF_Q    = ALG(OFF_SDEC+ (size_t)Bb*DEC*4);         // f32 [B][ATT]
static constexpr size_t OFF_H1   = ALG(OFF_Q   + (size_t)Bb*ATT*4);
static constexpr size_t OFF_C1   = ALG(OFF_H1  + (size_t)Bb*DEC*4);
static constexpr size_t OFF_H2   = ALG(OFF_C1  + (size_t)Bb*DEC*4);
static constexpr size_t OFF_C2   = ALG(OFF_H2  + (size_t)Bb*DEC*4);
static constexpr size_t OFF_CTX  = ALG(OFF_C2  + (size_t)Bb*DEC*4);         // f32 [2][B][ENC]
static constexpr size_t OFF_G1   = ALG(OFF_CTX + (size_t)2*Bb*ENC*4);       // f32 [B][Gg]
static constexpr size_t OFF_G2   = ALG(OFF_G1  + (size_t)Bb*Gg*4);
static constexpr size_t OFF_AF1  = ALG(OFF_G2  + (size_t)Bb*Gg*4);          // bf16 A-frags gemm1 [2][NKT1][32][16]
static constexpr size_t OFF_AF2  = ALG(OFF_AF1 + (size_t)2*NKT1*512*2);     // bf16 A-frags gemm2 [2][NKT2][32][16]
static constexpr size_t OFF_SYNC = ALG(OFF_AF2 + (size_t)2*NKT2*512*2);

struct P {
  const float *enc, *melT, *spk;
  const float *pw1, *pb1, *pw2, *pb2;
  const float *aqw, *aqb, *akw, *akb, *asw, *asb, *aew, *aeb;
  const float *l1wih, *l1whh, *l1bih, *l1bhh;
  const float *l2wih, *l2whh, *l2bih, *l2bhh;
  const float *skw, *skb, *melw, *melb, *gw, *gb;
  float* out;
  char*  ws;
};

__device__ __forceinline__ float sigf(float x){ return 1.0f/(1.0f + __expf(-x)); }

// deterministic dropout mask (p=0.5, inverted): returns 0.0 or 2.0
__device__ __forceinline__ float dropmask(unsigned t, unsigned layer, unsigned idx){
  unsigned h = t*0x9E3779B9u ^ (layer+1u)*0x85EBCA6Bu ^ idx*0xC2B2AE35u;
  h ^= h >> 16; h *= 0x7FEB352Du; h ^= h >> 15; h *= 0x846CA68Bu; h ^= h >> 16;
  return (h & 1u) ? 2.0f : 0.0f;
}

// device-wide barrier (32 blocks are always co-resident on CDNA5)
__device__ __forceinline__ void grid_sync(unsigned* sy){
  __threadfence();
  __syncthreads();
  if (threadIdx.x == 0){
    unsigned g = __hip_atomic_load(&sy[1], __ATOMIC_ACQUIRE, __HIP_MEMORY_SCOPE_AGENT);
    unsigned a = __hip_atomic_fetch_add(&sy[0], 1u, __ATOMIC_ACQ_REL, __HIP_MEMORY_SCOPE_AGENT);
    if (a == NWG - 1u){
      __hip_atomic_store(&sy[0], 0u, __ATOMIC_RELAXED, __HIP_MEMORY_SCOPE_AGENT);
      __hip_atomic_fetch_add(&sy[1], 1u, __ATOMIC_RELEASE, __HIP_MEMORY_SCOPE_AGENT);
    } else {
      while (__hip_atomic_load(&sy[1], __ATOMIC_ACQUIRE, __HIP_MEMORY_SCOPE_AGENT) == g)
        __builtin_amdgcn_s_sleep(1);
    }
  }
  __syncthreads();
  __threadfence();
}

// scatter one fp32 activation element (row, k) into the bf16 A-fragment buffer
// A-layout (ISA 16-bit A 16x32): lane = (row&15) | (koff&8 ? 16 : 0),
// element i = (koff&7) + (koff&16 ? 8 : 0), tile block mtile = row>>4
__device__ __forceinline__ void store_afrag(__bf16* __restrict__ af, int nkt, int row, int k, float v){
  int kt = k >> 5, koff = k & 31;
  int lane = (row & 15) | ((koff & 8) ? 16 : 0);
  int i = (koff & 7) + ((koff & 16) ? 8 : 0);
  int mt = row >> 4;
  af[(((size_t)mt*nkt + kt)*32 + lane)*16 + i] = (__bf16)v;
}

// load a pre-packed fragment: 16 contiguous bf16 per lane (two b128 loads)
__device__ __forceinline__ v16bf load_frag(const __bf16* __restrict__ fp){
  v16bf f;
#pragma unroll
  for (int i = 0; i < 16; ++i) f[i] = fp[i];
  return f;
}

// gates[32][4096] = Afrags x Wpacked   (one 32x16 strip per wave)
__device__ __forceinline__ void gemm_gates(const __bf16* __restrict__ wp,
                                           const __bf16* __restrict__ af,
                                           int nkt, float* __restrict__ gout,
                                           int lane, int wave){
  const int strip = wave;                 // 256 waves == 256 strips of 16 cols
  v8f acc0 = {}; v8f acc1 = {};
  const __bf16* bpp = wp + ((size_t)strip*nkt*32 + lane)*16;
  const __bf16* a0p = af + (size_t)lane*16;
  const __bf16* a1p = af + ((size_t)nkt*32 + lane)*16;
  for (int kt = 0; kt < nkt; ++kt){
    v16bf a0 = load_frag(a0p);
    v16bf a1 = load_frag(a1p);
    v16bf bm = load_frag(bpp);
    if (kt + 1 < nkt)
      __builtin_prefetch((const void*)(bpp + 512), 0, 0);   // global_prefetch_b8: next B tile (L2-resident)
    acc0 = __builtin_amdgcn_wmma_f32_16x16x32_bf16(false, a0, false, bm, (short)0, acc0, false, false);
    acc1 = __builtin_amdgcn_wmma_f32_16x16x32_bf16(false, a1, false, bm, (short)0, acc1, false, false);
    a0p += 512; a1p += 512; bpp += 512;
  }
  // C/D layout: VGPR r: lanes0-15 -> M=r, lanes16-31 -> M=8+r; N = strip*16 + (lane&15)
  const int col = strip*16 + (lane & 15);
  const int rb  = (lane & 16) ? 8 : 0;
#pragma unroll
  for (int r = 0; r < 8; ++r){
    gout[(size_t)(rb + r)*Gg + col]      = acc0[r];
    gout[(size_t)(16 + rb + r)*Gg + col] = acc1[r];
  }
}

// mel/gate output heads for step t_out; each 1536-dot split across a lane pair
__device__ __forceinline__ void emit_outputs(const P& p, const float* h2, const float* ctxp,
                                             int t_out, int gid,
                                             float* out_mel, float* out_gate){
  const int o2 = gid;                        // (Bb*MEL+Bb)*2 = 5184 = 162 full waves
  if (o2 < (Bb*MEL + Bb)*2){
    const int o = o2 >> 1, half = o2 & 1;
    const bool isgate = (o >= Bb*MEL);
    int bo, m; const float* w; float bias;
    if (!isgate){ bo = o/MEL; m = o%MEL; w = p.melw + m; bias = p.melb[m]; }
    else        { bo = o - Bb*MEL; m = 0; w = p.gw;      bias = p.gb[0];   }
    const float* hb = h2  + (size_t)bo*DEC;
    const float* cb = ctxp+ (size_t)bo*ENC;
    float acc = 0.f;
    const int k0 = half*768, k1 = k0 + 768;
    for (int k = k0; k < k1; ++k){
      float v  = (k < DEC) ? hb[k] : cb[k - DEC];
      float wv = isgate ? w[k] : w[(size_t)k*MEL];
      acc += v*wv;
    }
    acc += __shfl_xor(acc, 1);
    if (half == 0){
      if (!isgate) out_mel[((size_t)bo*Tt + t_out)*MEL + m] = acc + bias;
      else         out_gate[(size_t)bo*Tt + t_out]          = acc + bias;
    }
  }
}

__global__ void init_sync_kernel(unsigned* sy){ if (threadIdx.x < 2) sy[threadIdx.x] = 0u; }

__global__ __launch_bounds__(NT) void tts_decoder_kernel(P p){
  const int tid  = threadIdx.x;
  const int gid  = blockIdx.x*NT + tid;
  const int lane = tid & 31;
  const int wave = gid >> 5;

  char* ws = p.ws;
  __bf16* w1p = (__bf16*)(ws + OFF_W1P);
  __bf16* w2p = (__bf16*)(ws + OFF_W2P);
  float*  pk  = (float*)(ws + OFF_PK);
  float*  satt= (float*)(ws + OFF_SATT);
  float*  sdec= (float*)(ws + OFF_SDEC);
  float*  qb  = (float*)(ws + OFF_Q);
  float*  h1  = (float*)(ws + OFF_H1);
  float*  c1  = (float*)(ws + OFF_C1);
  float*  h2  = (float*)(ws + OFF_H2);
  float*  c2  = (float*)(ws + OFF_C2);
  float*  ctx = (float*)(ws + OFF_CTX);
  float*  g1  = (float*)(ws + OFF_G1);
  float*  g2  = (float*)(ws + OFF_G2);
  __bf16* af1 = (__bf16*)(ws + OFF_AF1);
  __bf16* af2 = (__bf16*)(ws + OFF_AF2);
  unsigned* sy= (unsigned*)(ws + OFF_SYNC);

  float* out_mel  = p.out;
  float* out_gate = p.out + (size_t)Bb*Tt*MEL;
  float* out_aw   = out_gate + (size_t)Bb*Tt;

  // ================= P1: one-time pack + hoisted projections =================
  // pack W1cat = [l1_wih ; l1_whh] into WMMA B-fragment layout (bf16)
  for (size_t d = gid; d < (size_t)NSTRIP*NKT1*512; d += GSZ){
    int i = (int)(d & 15), ln = (int)((d >> 4) & 31);
    size_t r = d >> 9;
    int kt = (int)(r % NKT1), strip = (int)(r / NKT1);
    int K = kt*32 + (i & 7) + ((ln & 16) ? 8 : 0) + ((i & 8) ? 16 : 0);
    int N = strip*16 + (ln & 15);
    float v = (K < 640) ? p.l1wih[(size_t)K*Gg + N] : p.l1whh[(size_t)(K - 640)*Gg + N];
    w1p[d] = (__bf16)v;
  }
  // pack W2cat = [l2_wih ; l2_whh]
  for (size_t d = gid; d < (size_t)NSTRIP*NKT2*512; d += GSZ){
    int i = (int)(d & 15), ln = (int)((d >> 4) & 31);
    size_t r = d >> 9;
    int kt = (int)(r % NKT2), strip = (int)(r / NKT2);
    int K = kt*32 + (i & 7) + ((ln & 16) ? 8 : 0) + ((i & 8) ? 16 : 0);
    int N = strip*16 + (ln & 15);
    float v = (K < DEC) ? p.l2wih[(size_t)K*Gg + N] : p.l2whh[(size_t)(K - DEC)*Gg + N];
    w2p[d] = (__bf16)v;
  }
  // proc_keys = enc @ att_k_w + b
  for (int o = gid; o < Bb*Ss*ATT; o += GSZ){
    int a = o & (ATT-1), bs = o >> 7;
    const float* e = p.enc + (size_t)bs*ENC;
    float acc = p.akb[a];
    for (int k = 0; k < ENC; ++k) acc += e[k]*p.akw[(size_t)k*ATT + a];
    pk[o] = acc;
  }
  // spk_att and q0 = att_q_b + spk_att (h1 starts at zero)
  for (int o = gid; o < Bb*ATT; o += GSZ){
    int a = o & (ATT-1), b = o >> 7;
    const float* s = p.spk + (size_t)b*SPK;
    float acc = p.asb[a];
    for (int k = 0; k < SPK; ++k) acc += s[k]*p.asw[(size_t)k*ATT + a];
    satt[o] = acc;
    qb[o]   = acc + p.aqb[a];
  }
  // spk_dec
  for (int o = gid; o < Bb*DEC; o += GSZ){
    int j = o & (DEC-1), b = o >> 10;
    const float* s = p.spk + (size_t)b*SPK;
    float acc = p.skb[j];
    for (int k = 0; k < SPK; ++k) acc += s[k]*p.skw[(size_t)k*DEC + j];
    sdec[o] = acc;
  }
  // zero recurrent state and the A-fragment buffers (h1/h2 regions must start at 0)
  for (int o = gid; o < Bb*DEC; o += GSZ){ h1[o]=0.f; c1[o]=0.f; h2[o]=0.f; c2[o]=0.f; }
  {
    unsigned* z1 = (unsigned*)af1;  // 2*NKT1*512 bf16 = 53248 -> 26624 dwords
    unsigned* z2 = (unsigned*)af2;  // 2*NKT2*512 bf16 = 65536 -> 32768 dwords
    for (int o = gid; o < 26624; o += GSZ) z1[o] = 0u;
    for (int o = gid; o < 32768; o += GSZ) z2[o] = 0u;
  }
  grid_sync(sy);

  __shared__ float sh_q[ATT], sh_ew[ATT], sh_h[PRE], sh_aw[Ss], sh_red[NT];

  for (int t = 0; t < Tt; ++t){
    // ========= Stage B: prenet + attention =========
    {
      const int b = blockIdx.x;                       // NWG == Bb
      if (tid < ATT){ sh_q[tid] = qb[b*ATT + tid]; sh_ew[tid] = p.aew[tid]; }
      // prenet layer1 (PRE == NT); teacher forcing: input at t==0 is zeros
      {
        float acc = p.pb1[tid];
        if (t > 0){
          const float* mi = p.melT + ((size_t)b*Tt + (t-1))*MEL;
          for (int m = 0; m < MEL; ++m) acc += mi[m]*p.pw1[(size_t)m*PRE + tid];
        }
        sh_h[tid] = fmaxf(acc, 0.f) * dropmask((unsigned)t, 0u, (unsigned)(b*PRE + tid));
      }
      __syncthreads();
      if (tid < PREOUT){
        float acc = p.pb2[tid];
        for (int k = 0; k < PRE; ++k) acc += sh_h[k]*p.pw2[(size_t)k*PREOUT + tid];
        float x = fmaxf(acc, 0.f) * dropmask((unsigned)t, 1u, (unsigned)(b*PREOUT + tid));
        store_afrag(af1, NKT1, b, tid, x);            // k = 0..127 of [x|ctx|h1]
      }
      // attention energy, one s per thread (Ss == NT)
      {
        const float* pkr = pk + ((size_t)b*Ss + tid)*ATT;
        float e = p.aeb[0];
        for (int a = 0; a < ATT; ++a) e += tanhf(sh_q[a] + pkr[a])*sh_ew[a];
        sh_aw[tid] = e; sh_red[tid] = e;
      }
      __syncthreads();
      for (int off = NT/2; off > 0; off >>= 1){
        if (tid < off) sh_red[tid] = fmaxf(sh_red[tid], sh_red[tid+off]);
        __syncthreads();
      }
      float mx = sh_red[0];
      __syncthreads();
      float ex = __expf(sh_aw[tid] - mx);
      sh_red[tid] = ex;
      __syncthreads();
      for (int off = NT/2; off > 0; off >>= 1){
        if (tid < off) sh_red[tid] += sh_red[tid+off];
        __syncthreads();
      }
      float aw = ex * (1.0f/sh_red[0]);
      __syncthreads();
      sh_aw[tid] = aw;
      out_aw[((size_t)b*Tt + t)*Ss + tid] = aw;
      __syncthreads();
      // ctx = aw @ enc  (512 cols over 256 threads)
      float* ctxc = ctx + (size_t)(t & 1)*Bb*ENC + (size_t)b*ENC;
      for (int j = tid; j < ENC; j += NT){
        float acc = 0.f;
        const float* e0 = p.enc + (size_t)b*Ss*ENC + j;
        for (int s = 0; s < Ss; ++s) acc += sh_aw[s]*e0[(size_t)s*ENC];
        ctxc[j] = acc;
        store_afrag(af1, NKT1, b, PREOUT + j, acc);   // k = 128..639
      }
    }
    grid_sync(sy);
    // ========= Stage C: gates1 = [x|ctx|h1] @ W1cat  (WMMA bf16) =========
    gemm_gates(w1p, af1, NKT1, g1, lane, wave);
    grid_sync(sy);
    // ========= Stage D: LSTM1 cell update (+ mel/gate outputs for t-1) =========
    for (int o = gid; o < Bb*DEC; o += GSZ){
      int j = o & (DEC-1), b = o >> 10;
      const float* g = g1 + (size_t)b*Gg;
      float gi = g[j]         + p.l1bih[j]         + p.l1bhh[j];
      float gf = g[DEC + j]   + p.l1bih[DEC + j]   + p.l1bhh[DEC + j];
      float gg = g[2*DEC + j] + p.l1bih[2*DEC + j] + p.l1bhh[2*DEC + j];
      float go = g[3*DEC + j] + p.l1bih[3*DEC + j] + p.l1bhh[3*DEC + j];
      float c  = sigf(gf)*c1[o] + sigf(gi)*tanhf(gg);
      float h  = sigf(go)*tanhf(c);
      c1[o] = c; h1[o] = h;
      store_afrag(af1, NKT1, b, 640 + j, h);          // gemm1 k = 640..1663
      store_afrag(af2, NKT2, b, j, h);                // gemm2 k = 0..1023
    }
    if (t > 0)
      emit_outputs(p, h2, ctx + (size_t)((t-1) & 1)*Bb*ENC, t-1, gid, out_mel, out_gate);
    grid_sync(sy);
    // ========= Stage E: gates2 = [h1|h2] @ W2cat (WMMA) ; q for step t+1 =========
    gemm_gates(w2p, af2, NKT2, g2, lane, wave);
    {
      // q split across lane pairs: GSZ == Bb*ATT*2
      const int o = gid >> 1, half = gid & 1;
      const int a = o & (ATT-1), b = o >> 7;
      const float* hb = h1 + (size_t)b*DEC + half*512;
      const float* wq = p.aqw + (size_t)(half*512)*ATT + a;
      float acc = 0.f;
      for (int k = 0; k < 512; ++k) acc += hb[k]*wq[(size_t)k*ATT];
      acc += __shfl_xor(acc, 1);
      if (half == 0) qb[o] = acc + p.aqb[a] + satt[o];
    }
    grid_sync(sy);
    // ========= Stage F: LSTM2 cell update (+ speaker projection carried in h2) =========
    for (int o = gid; o < Bb*DEC; o += GSZ){
      int j = o & (DEC-1), b = o >> 10;
      const float* g = g2 + (size_t)b*Gg;
      float gi = g[j]         + p.l2bih[j]         + p.l2bhh[j];
      float gf = g[DEC + j]   + p.l2bih[DEC + j]   + p.l2bhh[DEC + j];
      float gg = g[2*DEC + j] + p.l2bih[2*DEC + j] + p.l2bhh[2*DEC + j];
      float go = g[3*DEC + j] + p.l2bih[3*DEC + j] + p.l2bhh[3*DEC + j];
      float c  = sigf(gf)*c2[o] + sigf(gi)*tanhf(gg);
      float h  = sigf(go)*tanhf(c) + sdec[o];
      c2[o] = c; h2[o] = h;
      store_afrag(af2, NKT2, b, DEC + j, h);          // gemm2 k = 1024..2047
    }
    grid_sync(sy);
  }
  // ========= epilogue: outputs for final step =========
  emit_outputs(p, h2, ctx + (size_t)((Tt-1) & 1)*Bb*ENC, Tt-1, gid, out_mel, out_gate);
}

extern "C" void kernel_launch(void* const* d_in, const int* in_sizes, int n_in,
                              void* d_out, int out_size, void* d_ws, size_t ws_size,
                              hipStream_t stream){
  (void)in_sizes; (void)n_in; (void)out_size; (void)ws_size;
  P p;
  p.enc   = (const float*)d_in[0];
  p.melT  = (const float*)d_in[1];
  p.spk   = (const float*)d_in[2];
  p.pw1   = (const float*)d_in[3];  p.pb1 = (const float*)d_in[4];
  p.pw2   = (const float*)d_in[5];  p.pb2 = (const float*)d_in[6];
  p.aqw   = (const float*)d_in[7];  p.aqb = (const float*)d_in[8];
  p.akw   = (const float*)d_in[9];  p.akb = (const float*)d_in[10];
  p.asw   = (const float*)d_in[11]; p.asb = (const float*)d_in[12];
  p.aew   = (const float*)d_in[13]; p.aeb = (const float*)d_in[14];
  p.l1wih = (const float*)d_in[15]; p.l1whh = (const float*)d_in[16];
  p.l1bih = (const float*)d_in[17]; p.l1bhh = (const float*)d_in[18];
  p.l2wih = (const float*)d_in[19]; p.l2whh = (const float*)d_in[20];
  p.l2bih = (const float*)d_in[21]; p.l2bhh = (const float*)d_in[22];
  p.skw   = (const float*)d_in[23]; p.skb = (const float*)d_in[24];
  p.melw  = (const float*)d_in[25]; p.melb = (const float*)d_in[26];
  p.gw    = (const float*)d_in[27]; p.gb  = (const float*)d_in[28];
  p.out   = (float*)d_out;
  p.ws    = (char*)d_ws;

  init_sync_kernel<<<1, 64, 0, stream>>>((unsigned*)((char*)d_ws + OFF_SYNC));
  tts_decoder_kernel<<<NWG, NT, 0, stream>>>(p);
}
